// AttentionBlock_14267881357776
// MI455X (gfx1250) — matrix operand order repaired and checked by reading, not compiled, not run on hardware
//
#include <hip/hip_runtime.h>
#include <hip/hip_bf16.h>
#include <math.h>

// ---------------------------------------------------------------------------
// Problem constants (match reference)
// ---------------------------------------------------------------------------
#define BB   16
#define TQ   2048
#define TK   2048
#define CC   256   // NUM_CHANNELS
#define AA   256   // ATTN_SIZE
#define EE   256   // EMBED_SIZE

// LDS padded strides (in halves).  Bank period = 64 banks * 4B = 256B.
// 264 halves = 528B = 132 dwords ; 132 mod 64 = 4  -> lanes 0..15 hit 16 distinct banks
// 40 halves  = 80B  = 20 dwords  ; gcd(20,64) = 4  -> lanes 0..15 hit 16 distinct banks
#define KSTRIDE 264
#define VSTRIDE 40
#define PSTRIDE 40
#define CSTRIDE 264

typedef __attribute__((ext_vector_type(16))) _Float16 v16h;
typedef __attribute__((ext_vector_type(8)))  _Float16 v8h;
typedef __attribute__((ext_vector_type(8)))  float    v8f;

__device__ __forceinline__ v8f wmma_f16(v16h a, v16h b, v8f c) {
    // D = A(16x32 f16) * B(32x16 f16) + C(16x16 f32)
    return __builtin_amdgcn_wmma_f32_16x16x32_f16(
        /*neg_a=*/false, a, /*neg_b=*/false, b,
        /*c_mod=*/(short)0, c, /*reuse_a=*/false, /*reuse_b=*/false);
}

__device__ __forceinline__ v16h cat8(v8h lo, v8h hi) {
    return __builtin_shufflevector(lo, hi, 0,1,2,3,4,5,6,7,8,9,10,11,12,13,14,15);
}

__device__ __forceinline__ v16h ld_frag(const _Float16* p) {
    // 32B fragment as two 16B-aligned LDS/global loads
    v8h lo = *(const v8h*)p;
    v8h hi = *(const v8h*)(p + 8);
    return cat8(lo, hi);
}

// Low 32 bits of a generic LDS pointer == wave-relative LDS byte offset.
__device__ __forceinline__ unsigned ldsoff(const void* p) {
    return (unsigned)(size_t)(const __attribute__((address_space(3))) void*)p;
}

// CDNA5 async DMA: 16B per lane, global -> LDS, tracked by ASYNCcnt.
__device__ __forceinline__ void async_copy16(void* lds_dst, const _Float16* gsrc) {
    unsigned l = ldsoff(lds_dst);
    asm volatile("global_load_async_to_lds_b128 %0, %1, off"
                 :: "v"(l), "v"(gsrc) : "memory");
}

__device__ __forceinline__ void wait_async0() {
    asm volatile("s_wait_asynccnt 0" ::: "memory");
}

// pe(t, col) = sin/cos( t^2 / 10000^{(2*(col//2))/C} ), rate == 1.0 for both Q & K
__device__ __forceinline__ float pe_val(int t, int col) {
    float tt  = (float)t;
    float num = tt * tt;
    float e   = (float)(col & ~1) * (13.287712379549449f / 256.0f);  // log2(1e4)/256
    float ang = num * exp2f(-e);
    return (col & 1) ? __cosf(ang) : __sinf(ang);
}

// ---------------------------------------------------------------------------
// Kernel 0: weight transpose + f32 -> f16 convert.  Wt[a][c] = W[c][a]
// ---------------------------------------------------------------------------
__global__ __launch_bounds__(256)
void wprep(const float* __restrict__ W, _Float16* __restrict__ Wt) {
    int idx = blockIdx.x * 256 + threadIdx.x;     // 0 .. 65535
    int a = idx & 255;
    int c = idx >> 8;
    Wt[(size_t)a * CC + c] = (_Float16)W[(size_t)c * AA + a];
}

// ---------------------------------------------------------------------------
// Kernel 1: projection  dst = (X [+PE]) @ W + b   via WMMA f16
// Block = 128 threads = 4 waves; each wave produces 16 rows x 256 cols.
// ---------------------------------------------------------------------------
__global__ __launch_bounds__(128)
void proj(const float* __restrict__ X, const _Float16* __restrict__ Wt,
          const float* __restrict__ bias, _Float16* __restrict__ dst,
          int T, int addPE, int storeT)
{
    const int lane = threadIdx.x & 31;
    const int wave = threadIdx.x >> 5;
    const int hf   = lane >> 4;
    const int lm   = lane & 15;
    const int b    = blockIdx.y;
    const int t0   = blockIdx.x * 64 + wave * 16;
    const int row  = t0 + lm;

    // --- Build 8 A-fragments covering the 256-wide K (input-channel) dim ---
    v16h afrag[8];
    const float* xrow = X + ((size_t)b * T + row) * CC;
    #pragma unroll
    for (int c = 0; c < 8; ++c) {
        const int base = c * 32 + hf * 16;
        union { v16h v; _Float16 h[16]; } u;
        #pragma unroll
        for (int h = 0; h < 16; ++h) {
            int col = base + h;   // contiguous K layout: k = 16*hf + h
            float v = xrow[col];
            if (addPE) v += pe_val(row, col);
            u.h[h] = (_Float16)v;
        }
        afrag[c] = u.v;
    }

    // --- 16 output N-tiles of 16, accumulate over 8 K-chunks ---
    const v16h* wv = (const v16h*)Wt;
    v8f acc[16];
    #pragma unroll
    for (int t = 0; t < 16; ++t) acc[t] = (v8f){};
    #pragma unroll
    for (int t = 0; t < 16; ++t) {
        const size_t rbase = (size_t)(t * 16 + lm) * 16;
        #pragma unroll
        for (int c = 0; c < 8; ++c)
            acc[t] = wmma_f16(afrag[c], wv[rbase + c * 2 + hf], acc[t]);
    }

    // --- bias + store (f16) ---
    #pragma unroll
    for (int t = 0; t < 16; ++t) {
        const int   col = t * 16 + lm;
        const float bv  = bias[col];
        #pragma unroll
        for (int i = 0; i < 8; ++i) {
            const int r = t0 + i + 8 * hf;
            _Float16 o = (_Float16)(acc[t][i] + bv);
            if (storeT) dst[((size_t)b * AA + col) * T + r] = o;   // V^T layout
            else        dst[((size_t)b * T + r) * AA + col] = o;
        }
    }
}

// ---------------------------------------------------------------------------
// Cooperative async staging of one 32-key step: K tile (32x256 halves) and
// V^T tile (256a x 32k halves) into padded LDS.  128 threads x 16B x 16 issues.
// ---------------------------------------------------------------------------
__device__ __forceinline__ void issue_tile_loads(
    const _Float16* __restrict__ Kg, const _Float16* __restrict__ Vg,
    int b, int k0, _Float16* sKbuf, _Float16* sVbuf, int tid)
{
    // K tile: 32 rows x 512B = 1024 chunks of 16B
    #pragma unroll
    for (int j = 0; j < 8; ++j) {
        int e   = tid + j * 128;
        int r   = e >> 5;              // key row within tile
        int c16 = e & 31;              // 16B chunk within 512B row
        const _Float16* g = Kg + (size_t)(b * TK + k0 + r) * AA + c16 * 8;
        async_copy16(sKbuf + r * KSTRIDE + c16 * 8, g);
    }
    // V^T tile: 256 a-rows x 64B = 1024 chunks of 16B
    #pragma unroll
    for (int j = 0; j < 8; ++j) {
        int e  = tid + j * 128;
        int a  = e >> 2;               // feature row
        int ck = e & 3;                // 8-half chunk within 32 k
        const _Float16* g = Vg + (size_t)(b * AA + a) * TK + k0 + ck * 8;
        async_copy16(sVbuf + a * VSTRIDE + ck * 8, g);
    }
}

// ---------------------------------------------------------------------------
// Kernel 2: flash attention + fused output projection.
// 128 threads = 4 waves; K/V tiles double-buffered in LDS via async DMA,
// shared by all 4 waves.  Each wave owns 16 query rows.
// ---------------------------------------------------------------------------
__global__ __launch_bounds__(128)
void attn(const _Float16* __restrict__ Q, const _Float16* __restrict__ K,
          const _Float16* __restrict__ Vt, const _Float16* __restrict__ WoT,
          const float* __restrict__ bo, float* __restrict__ out)
{
    __shared__ _Float16 sK[2][32 * KSTRIDE];    // 33,792 B (reused as ctx stage)
    __shared__ _Float16 sV[2][256 * VSTRIDE];   // 40,960 B
    __shared__ _Float16 sP[4][16 * PSTRIDE];    //  5,120 B

    const int tid  = threadIdx.x;
    const int lane = tid & 31;
    const int wave = tid >> 5;
    const int hf   = lane >> 4;
    const int lm   = lane & 15;
    const int b    = blockIdx.y;
    const int q0   = blockIdx.x * 64 + wave * 16;

    // --- Q A-fragments (16 rows x 256 features), loaded once from global ---
    v16h qfrag[8];
    const _Float16* qrow = Q + ((size_t)b * TQ + q0 + lm) * AA;
    #pragma unroll
    for (int c = 0; c < 8; ++c) {
        v8h lo = *(const v8h*)(qrow + c * 32 + hf * 16);
        v8h hi = *(const v8h*)(qrow + c * 32 + hf * 16 + 8);
        qfrag[c] = cat8(lo, hi);
    }

    v8f acc[16];
    #pragma unroll
    for (int t = 0; t < 16; ++t) acc[t] = (v8f){};
    float m[8], l[8];
    #pragma unroll
    for (int i = 0; i < 8; ++i) { m[i] = -INFINITY; l[i] = 0.0f; }

    // --- prologue: stage first tile ---
    issue_tile_loads(K, Vt, b, 0, sK[0], sV[0], tid);
    wait_async0();
    __syncthreads();

    int cur = 0;
    for (int k0 = 0; k0 < TK; k0 += 32) {
        // prefetch next tile into the other buffer while computing on this one
        if (k0 + 32 < TK)
            issue_tile_loads(K, Vt, b, k0 + 32, sK[cur ^ 1], sV[cur ^ 1], tid);

        const _Float16* kb = sK[cur];
        const _Float16* vb = sV[cur];

        // ---- scores for 32 keys: two 16x16 tiles (B-frags from LDS) ----
        v8f s0 = (v8f){}, s1 = (v8f){};
        #pragma unroll
        for (int c = 0; c < 8; ++c) {
            v16h kf0 = ld_frag(kb + lm        * KSTRIDE + c * 32 + hf * 16);
            v16h kf1 = ld_frag(kb + (16 + lm) * KSTRIDE + c * 32 + hf * 16);
            s0 = wmma_f16(qfrag[c], kf0, s0);
            s1 = wmma_f16(qfrag[c], kf1, s1);
        }

        // ---- online softmax (rows live in groups of 16 lanes) ----
        float mc[8];
        #pragma unroll
        for (int i = 0; i < 8; ++i) mc[i] = fmaxf(s0[i], s1[i]);
        #pragma unroll
        for (int d = 1; d < 16; d <<= 1)
            #pragma unroll
            for (int i = 0; i < 8; ++i) mc[i] = fmaxf(mc[i], __shfl_xor(mc[i], d, 32));

        float sc[8], rs[8];
        #pragma unroll
        for (int i = 0; i < 8; ++i) {
            float mn = fmaxf(m[i], mc[i]);
            sc[i] = __expf(m[i] - mn);
            m[i]  = mn;
            float p0 = __expf(s0[i] - mn);
            float p1 = __expf(s1[i] - mn);
            s0[i] = p0; s1[i] = p1;
            rs[i] = p0 + p1;
        }
        #pragma unroll
        for (int d = 1; d < 16; d <<= 1)
            #pragma unroll
            for (int i = 0; i < 8; ++i) rs[i] += __shfl_xor(rs[i], d, 32);
        #pragma unroll
        for (int i = 0; i < 8; ++i) l[i] = l[i] * sc[i] + rs[i];
        #pragma unroll
        for (int t = 0; t < 16; ++t)
            #pragma unroll
            for (int i = 0; i < 8; ++i) acc[t][i] = acc[t][i] * sc[i];

        // ---- stage P (16x32) into this wave's LDS slab: C-layout -> A-frag ----
        _Float16* pslab = sP[wave];
        #pragma unroll
        for (int i = 0; i < 8; ++i) {
            int r = i + 8 * hf;
            pslab[r * PSTRIDE + lm]      = (_Float16)s0[i];
            pslab[r * PSTRIDE + 16 + lm] = (_Float16)s1[i];
        }
        v8h plo = *(const v8h*)(pslab + lm * PSTRIDE + hf * 16);
        v8h phi = *(const v8h*)(pslab + lm * PSTRIDE + hf * 16 + 8);
        v16h pf = cat8(plo, phi);

        // ---- context += P @ V  (B-frags from LDS V^T tile) ----
        #pragma unroll
        for (int t = 0; t < 16; ++t) {
            v16h vf = ld_frag(vb + (t * 16 + lm) * VSTRIDE + hf * 16);
            acc[t] = wmma_f16(pf, vf, acc[t]);
        }

        wait_async0();        // next tile landed
        __syncthreads();      // everyone done reading cur; safe to overwrite next iter
        cur ^= 1;
    }

    // ---- normalize, scale by rsqrt(TK), stage context (f16) in per-wave slab ----
    // (reuse the sK double-buffer region: 4 waves x 16 x CSTRIDE halves == sizeof sK)
    const float rsc = rsqrtf((float)TK);
    _Float16* ctx = ((_Float16*)sK) + (size_t)wave * 16 * CSTRIDE;
    #pragma unroll
    for (int t = 0; t < 16; ++t) {
        #pragma unroll
        for (int i = 0; i < 8; ++i) {
            int r = i + 8 * hf;
            ctx[r * CSTRIDE + t * 16 + lm] = (_Float16)(acc[t][i] / l[i] * rsc);
        }
    }

    // ---- fused output projection: out = ctx @ Wo + bo ----
    const v16h* wv = (const v16h*)WoT;
    v8f oacc[16];
    #pragma unroll
    for (int t = 0; t < 16; ++t) oacc[t] = (v8f){};
    #pragma unroll
    for (int c = 0; c < 8; ++c) {
        v8h lo = *(const v8h*)(ctx + lm * CSTRIDE + c * 32 + hf * 16);
        v8h hi = *(const v8h*)(ctx + lm * CSTRIDE + c * 32 + hf * 16 + 8);
        v16h cf = cat8(lo, hi);
        #pragma unroll
        for (int t = 0; t < 16; ++t)
            oacc[t] = wmma_f16(cf, wv[(size_t)(t * 16 + lm) * 16 + c * 2 + hf], oacc[t]);
    }
    #pragma unroll
    for (int t = 0; t < 16; ++t) {
        const int   col = t * 16 + lm;
        const float bv  = bo[col];
        #pragma unroll
        for (int i = 0; i < 8; ++i) {
            const int r = q0 + i + 8 * hf;
            out[((size_t)b * TQ + r) * EE + col] = oacc[t][i] + bv;
        }
    }
}

// ---------------------------------------------------------------------------
// Host-side launch
// ---------------------------------------------------------------------------
extern "C" void kernel_launch(void* const* d_in, const int* in_sizes, int n_in,
                              void* d_out, int out_size, void* d_ws, size_t ws_size,
                              hipStream_t stream) {
    const float* query  = (const float*)d_in[0];
    const float* keys   = (const float*)d_in[1];
    const float* values = (const float*)d_in[2];
    const float* W1 = (const float*)d_in[3];  const float* b1 = (const float*)d_in[4];
    const float* W2 = (const float*)d_in[5];  const float* b2 = (const float*)d_in[6];
    const float* W3 = (const float*)d_in[7];  const float* b3 = (const float*)d_in[8];
    const float* Wo = (const float*)d_in[9];  const float* bo = (const float*)d_in[10];
    float* out = (float*)d_out;

    _Float16* ws = (_Float16*)d_ws;
    const size_t QH = (size_t)BB * TQ * AA;        // halves per projected tensor
    _Float16* wsQ  = ws;
    _Float16* wsK  = ws + QH;
    _Float16* wsV  = ws + 2 * QH;                  // stored transposed [B][A][Tk]
    _Float16* wtW1 = ws + 3 * QH;
    _Float16* wtW2 = wtW1 + (size_t)CC * AA;
    _Float16* wtW3 = wtW2 + (size_t)CC * AA;
    _Float16* wtWo = wtW3 + (size_t)CC * AA;

    wprep<<<256, 256, 0, stream>>>(W1, wtW1);
    wprep<<<256, 256, 0, stream>>>(W2, wtW2);
    wprep<<<256, 256, 0, stream>>>(W3, wtW3);
    wprep<<<256, 256, 0, stream>>>(Wo, wtWo);

    dim3 grid(TQ / 64, BB), blk(128);
    proj<<<grid, blk, 0, stream>>>(query,  wtW2, b2, wsQ, TQ, /*addPE=*/1, /*storeT=*/0);
    proj<<<grid, blk, 0, stream>>>(keys,   wtW1, b1, wsK, TK, /*addPE=*/1, /*storeT=*/0);
    proj<<<grid, blk, 0, stream>>>(values, wtW3, b3, wsV, TK, /*addPE=*/0, /*storeT=*/1);

    attn<<<grid, blk, 0, stream>>>(wsQ, wsK, wsV, wtWo, bo, out);
}
